// ProbSparse_88210038326109
// MI455X (gfx1250) — compile-verified, matrix operand order checked by reading, not run on hardware
//
#include <hip/hip_runtime.h>
#include <hip/hip_bf16.h>
#include <math.h>

// ---------------- problem constants ----------------
#define Bx    16
#define Lx    960
#define Fx    7
#define Dx    512
#define Hx    8
#define DKx   64
#define UTOP  34
#define USAMP 960
#define LTILES 60              // L/16
#define MROWS  (Bx*Lx)         // 15360
#define MTILES (MROWS/16)      // 960
#define NTILES (Dx/16)         // 32
#define KC_CONV 48             // 1536/32
#define KC_QKV  16             // 512/32

typedef __attribute__((ext_vector_type(16))) _Float16 v16h;
typedef __attribute__((ext_vector_type(8)))  _Float16 v8h_t;   // one 16-byte fragment half
typedef __attribute__((ext_vector_type(8)))  float    v8f;
typedef __attribute__((ext_vector_type(4)))  unsigned int v4u;
typedef __attribute__((ext_vector_type(4)))  int      v4i;
typedef __attribute__((ext_vector_type(8)))  int      v8i;

union AFrag { v16h v; _Float16 e[16]; v8h_t h8[2]; };
union CFrag { v8f  v; float    e[8];  };
union HFrag { v8h_t v; _Float16 e[8]; };

#if __has_builtin(__builtin_amdgcn_tensor_load_to_lds) && __has_builtin(__builtin_amdgcn_s_wait_tensorcnt)
#define HAVE_TDM 1
#endif

// K-index pattern for 16-bit A/B fragments (16x32), ISA 7.12.2:
// lane r (r<16) holds K {0..7,16..23} (pairs), lane r+16 holds K {8..15,24..31}.
// => per lane the fragment is two contiguous 8-half (16B) runs at K = hi*8 and 16+hi*8.
__device__ __forceinline__ int kpat(int v, int hi) {
  return ((v & 3) * 2) + (hi ? 8 : 0) + ((v >= 4) ? 16 : 0);
}

#define WMMA_F16(Cv, Av, Bv) \
  __builtin_amdgcn_wmma_f32_16x16x32_f16(false, (Av), false, (Bv), (short)0, (Cv), false, false)

// load a 16x32 A/B fragment whose K-axis is contiguous in memory at `base`
__device__ __forceinline__ v16h frag_row(const _Float16* base, int hi) {
  AFrag a;
  a.h8[0] = *(const v8h_t*)(base + hi * 8);
  a.h8[1] = *(const v8h_t*)(base + 16 + hi * 8);
  return a.v;
}

// ---------------- 0: pack weight matrix (Kdim x 512, f32) into B-fragment layout ----------------
// Bp[((nt*KC + kc)*32 + lane)*16 + i]  ==  fragment element i for that lane.
__global__ void k_pack_b(const float* __restrict__ W, int Kdim, _Float16* __restrict__ Bp) {
  int idx = blockIdx.x * blockDim.x + threadIdx.x;
  int kcN = Kdim >> 5;
  int total = NTILES * kcN * 32;
  if (idx >= total) return;
  int lane = idx & 31;
  int kc = (idx >> 5) % kcN;
  int nt = (idx >> 5) / kcN;
  int r = lane & 15, hi = lane >> 4;
  int colg = nt * 16 + r;
  _Float16* dst = Bp + (size_t)idx * 16;
#pragma unroll
  for (int v2 = 0; v2 < 8; ++v2) {
    int kb = kc * 32 + kpat(v2, hi);
    dst[2 * v2]     = (_Float16)W[(size_t)kb * Dx + colg];
    dst[2 * v2 + 1] = (_Float16)W[(size_t)(kb + 1) * Dx + colg];
  }
}

// ---------------- 1: embedding ----------------
__global__ void k_embed(const float* __restrict__ values, const int* __restrict__ times,
                        const float* __restrict__ w_val, const float* __restrict__ b_val,
                        const float* __restrict__ pos_emb, const float* __restrict__ hour_tab,
                        const float* __restrict__ week_tab, const float* __restrict__ month_tab,
                        _Float16* __restrict__ emb16) {
  int idx = blockIdx.x * blockDim.x + threadIdx.x;
  if (idx >= MROWS * Dx) return;
  int d  = idx & (Dx - 1);
  int bl = idx >> 9;
  int l  = bl % Lx;
  float acc = b_val[d] + pos_emb[l * Dx + d];
  const float* vrow = values + bl * Fx;
#pragma unroll
  for (int f = 0; f < Fx; ++f) acc += vrow[f] * w_val[f * Dx + d];
  int t0 = times[bl * 3 + 0], t1 = times[bl * 3 + 1], t2 = times[bl * 3 + 2];
  acc += hour_tab[t0 * Dx + d] + week_tab[t1 * Dx + d] + month_tab[t2 * Dx + d];
  emb16[idx] = (_Float16)acc;
}

// ---------------- 2: conv1d(k=3, SAME) + bias + ELU via WMMA ----------------
// grid = (960); 8 waves/block; each wave owns 4 N-tiles of one 16-row M-tile.
__global__ void k_conv_elu(const _Float16* __restrict__ emb16,
                           const _Float16* __restrict__ convpack,
                           const float* __restrict__ conv_b,
                           _Float16* __restrict__ x16) {
  int wave = threadIdx.x >> 5;
  int lane = threadIdx.x & 31;
  int r = lane & 15, hi = lane >> 4;
  int mt = blockIdx.x;
  int nt0 = wave * 4;
  int row = mt * 16 + r;
  int b = row / Lx, l = row % Lx;
  const _Float16* arow = emb16 + (size_t)(b * Lx) * Dx;

  CFrag c[4];
#pragma unroll
  for (int tt = 0; tt < 4; ++tt)
#pragma unroll
    for (int j = 0; j < 8; ++j) c[tt].e[j] = 0.f;

  for (int kc = 0; kc < KC_CONV; ++kc) {
    int tap  = kc >> 4;                 // 0..2 (chunk never crosses a tap: 512%32==0)
    int dinb = (kc & 15) << 5;
    int l2   = l + tap - 1;             // SAME padding
    AFrag a;
    if (l2 >= 0 && l2 < Lx) {
      const _Float16* p = arow + (size_t)l2 * Dx + dinb;
      a.h8[0] = *(const v8h_t*)(p + hi * 8);
      a.h8[1] = *(const v8h_t*)(p + 16 + hi * 8);
    } else {
#pragma unroll
      for (int j = 0; j < 16; ++j) a.e[j] = (_Float16)0.f;
    }
#pragma unroll
    for (int tt = 0; tt < 4; ++tt) {
      const v8h_t* bp = (const v8h_t*)(convpack +
          ((((size_t)(nt0 + tt)) * KC_CONV + kc) * 32 + lane) * 16);
      AFrag bb;
      bb.h8[0] = bp[0];
      bb.h8[1] = bp[1];
      c[tt].v = WMMA_F16(c[tt].v, a.v, bb.v);
    }
  }
#pragma unroll
  for (int tt = 0; tt < 4; ++tt) {
    int colg = (nt0 + tt) * 16 + r;
    float bias = conv_b[colg];
#pragma unroll
    for (int j = 0; j < 8; ++j) {
      int orow = mt * 16 + j + hi * 8;
      float v = c[tt].e[j] + bias;
      v = v > 0.f ? v : (__expf(v) - 1.f);           // ELU
      x16[(size_t)orow * Dx + colg] = (_Float16)v;
    }
  }
}

// ---------------- 3: Q/K/V projection via WMMA ----------------
// grid = (960, 1, 3); z selects W/bias/output; 4 N-tiles per wave.
__global__ void k_qkv(const _Float16* __restrict__ x16, const _Float16* __restrict__ wpack,
                      const float* __restrict__ bq, const float* __restrict__ bk,
                      const float* __restrict__ bv,
                      _Float16* __restrict__ Q16, _Float16* __restrict__ K16,
                      float* __restrict__ Vf) {
  int which = blockIdx.z;
  const _Float16* Wp = wpack + (size_t)which * Dx * Dx;
  const float* bias = (which == 0) ? bq : ((which == 1) ? bk : bv);
  int wave = threadIdx.x >> 5, lane = threadIdx.x & 31;
  int r = lane & 15, hi = lane >> 4;
  int mt = blockIdx.x;
  int nt0 = wave * 4;
  int row = mt * 16 + r;
  const _Float16* xrow = x16 + (size_t)row * Dx;

  CFrag c[4];
#pragma unroll
  for (int tt = 0; tt < 4; ++tt)
#pragma unroll
    for (int j = 0; j < 8; ++j) c[tt].e[j] = 0.f;

  for (int kc = 0; kc < KC_QKV; ++kc) {
    AFrag a;
    const _Float16* p = xrow + kc * 32;
    __builtin_prefetch(p + 32, 0, 1);
    a.h8[0] = *(const v8h_t*)(p + hi * 8);
    a.h8[1] = *(const v8h_t*)(p + 16 + hi * 8);
#pragma unroll
    for (int tt = 0; tt < 4; ++tt) {
      const v8h_t* bp = (const v8h_t*)(Wp +
          ((((size_t)(nt0 + tt)) * KC_QKV + kc) * 32 + lane) * 16);
      AFrag bb;
      bb.h8[0] = bp[0];
      bb.h8[1] = bp[1];
      c[tt].v = WMMA_F16(c[tt].v, a.v, bb.v);
    }
  }
#pragma unroll
  for (int tt = 0; tt < 4; ++tt) {
    int colg = (nt0 + tt) * 16 + r;
    int h = colg >> 6, dk = colg & 63;
    float bs = bias[colg];
#pragma unroll
    for (int j = 0; j < 8; ++j) {
      int orow = mt * 16 + j + hi * 8;
      int b = orow / Lx, l = orow % Lx;
      size_t oi = (((size_t)(b * Hx + h)) * Lx + l) * DKx + dk;
      float v = c[tt].e[j] + bs;
      if (which == 0)      Q16[oi] = (_Float16)v;
      else if (which == 1) K16[oi] = (_Float16)v;
      else                 Vf[oi]  = v;
    }
  }
}

// ---------------- 4: M = max_m(Q.K_s) - mean_m(Q.K_s) via WMMA ----------------
// one wave per (b,h,ltile): 960 blocks x 8 waves.
__global__ void k_msparse(const _Float16* __restrict__ Q16, const _Float16* __restrict__ K16,
                          const int* __restrict__ sample_idx, float* __restrict__ Mbuf) {
  int wave = threadIdx.x >> 5, lane = threadIdx.x & 31;
  int wid = blockIdx.x * 8 + wave;
  int r = lane & 15, hi = lane >> 4;
  int bh = wid / LTILES;
  int lt = wid % LTILES;
  const _Float16* Qb = Q16 + (size_t)bh * Lx * DKx;
  const _Float16* Kb = K16 + (size_t)bh * Lx * DKx;

  const _Float16* qp = Qb + (size_t)(lt * 16 + r) * DKx;
  v16h a0 = frag_row(qp, hi);
  v16h a1 = frag_row(qp + 32, hi);

  float rmax[8], rsum[8];
#pragma unroll
  for (int j = 0; j < 8; ++j) { rmax[j] = -3.402823466e38f; rsum[j] = 0.f; }

  for (int mt = 0; mt < LTILES; ++mt) {
    int sj = sample_idx[mt * 16 + r];                 // gathered key column
    const _Float16* kp = Kb + (size_t)sj * DKx;
    __builtin_prefetch(kp, 0, 1);
    v16h b0 = frag_row(kp, hi);
    v16h b1 = frag_row(kp + 32, hi);
    CFrag c;
#pragma unroll
    for (int j = 0; j < 8; ++j) c.e[j] = 0.f;
    c.v = WMMA_F16(c.v, a0, b0);
    c.v = WMMA_F16(c.v, a1, b1);
#pragma unroll
    for (int j = 0; j < 8; ++j) { rmax[j] = fmaxf(rmax[j], c.e[j]); rsum[j] += c.e[j]; }
  }

#pragma unroll
  for (int j = 0; j < 8; ++j) {
    float mx = rmax[j], sm = rsum[j];
#pragma unroll
    for (int off = 1; off < 16; off <<= 1) {
      mx = fmaxf(mx, __shfl_xor(mx, off, 16));
      sm += __shfl_xor(sm, off, 16);
    }
    if (r == 0) {
      int rowl = lt * 16 + j + hi * 8;
      Mbuf[(size_t)bh * Lx + rowl] = mx - sm * (1.f / USAMP);
    }
  }
}

// ---------------- 5: top-34 selection per (b,h) ----------------
// M row staged into LDS with the Tensor Data Mover (1-D D# descriptor) when available.
__global__ void k_topk(const float* __restrict__ Mbuf, int* __restrict__ top_idx) {
  __shared__ float vals[Lx];
  __shared__ float rmx[256];
  __shared__ int   rix[256];
  int bh = blockIdx.x, t = threadIdx.x;
#ifdef HAVE_TDM
  if (t < 32) {                      // wave 0 issues the TDM transfer: 960 f32 -> LDS
    unsigned long long ga = (unsigned long long)(const void*)(Mbuf + (size_t)bh * Lx);
    unsigned ldsoff = (unsigned)(size_t)(void*)vals;
    v4u g0;
    g0.x = 1u;                                        // count=1, user descriptor
    g0.y = ldsoff;                                    // lds_addr
    g0.z = (unsigned)(ga & 0xFFFFFFFFu);              // global_addr[31:0]
    g0.w = (unsigned)((ga >> 32) & 0x1FFFFFFu) | (2u << 30);  // global_addr[56:32] | type=2
    v8i g1;
    g1[0] = (int)(2u << 16);     // workgroup_mask=0, data_size=4B
    g1[1] = (int)(960u << 16);   // tensor_dim0 = 960 (bits [63:48] of group1)
    g1[2] = (int)(1u << 16);     // tensor_dim0 hi=0, tensor_dim1 = 1
    g1[3] = (int)(960u << 16);   // tensor_dim1 hi=0, tile_dim0 = 960
    g1[4] = 0;                   // tile_dim1 = 0 (unused), tile_dim2 = 0
    g1[5] = 960;                 // tensor_dim0_stride
    g1[6] = 0;
    g1[7] = 0;
    v4i g2 = {0, 0, 0, 0};
    v4i g3 = {0, 0, 0, 0};
    v8i g4 = {0, 0, 0, 0, 0, 0, 0, 0};   // 6-arg toolchain form: extra group
    __builtin_amdgcn_tensor_load_to_lds(g0, g1, g2, g3, g4, 0);
    __builtin_amdgcn_s_wait_tensorcnt(0);
  }
  __syncthreads();
#else
  for (int i = t; i < Lx; i += 256) vals[i] = Mbuf[(size_t)bh * Lx + i];
  __syncthreads();
#endif
  for (int u = 0; u < UTOP; ++u) {
    float mx = -3.402823466e38f; int mi = Lx;
    for (int i = t; i < Lx; i += 256) {
      float v = vals[i];
      if (v > mx || (v == mx && i < mi)) { mx = v; mi = i; }
    }
    rmx[t] = mx; rix[t] = mi;
    __syncthreads();
    for (int s = 128; s > 0; s >>= 1) {
      if (t < s) {
        float v2 = rmx[t + s]; int i2 = rix[t + s];
        if (v2 > rmx[t] || (v2 == rmx[t] && i2 < rix[t])) { rmx[t] = v2; rix[t] = i2; }
      }
      __syncthreads();
    }
    if (t == 0) { top_idx[bh * UTOP + u] = rix[0]; vals[rix[0]] = -3.402823466e38f; }
    __syncthreads();
  }
}

// ---------------- 6: attention on top rows (softmax(QK/8) @ V) ----------------
__global__ void k_topattn(const _Float16* __restrict__ Q16, const _Float16* __restrict__ K16,
                          const float* __restrict__ Vf, const int* __restrict__ top_idx,
                          float* __restrict__ out_top) {
  __shared__ float q[DKx];
  __shared__ float sc[Lx];
  __shared__ float red[256];
  int blk = blockIdx.x;
  int bh = blk / UTOP, u = blk % UTOP;
  int t = threadIdx.x;
  int li = top_idx[bh * UTOP + u];
  const _Float16* Qr = Q16 + ((size_t)bh * Lx + li) * DKx;
  if (t < DKx) q[t] = (float)Qr[t];
  __syncthreads();
  const _Float16* Kb = K16 + (size_t)bh * Lx * DKx;
  float lmax = -3.402823466e38f;
  for (int n = t; n < Lx; n += 256) {
    const v8h_t* kr8 = (const v8h_t*)(Kb + (size_t)n * DKx);
    float s = 0.f;
#pragma unroll
    for (int cch = 0; cch < 8; ++cch) {
      HFrag kv; kv.v = kr8[cch];
#pragma unroll
      for (int j = 0; j < 8; ++j) s += q[cch * 8 + j] * (float)kv.e[j];
    }
    s *= 0.125f;                                      // 1/sqrt(DK)
    sc[n] = s;
    lmax = fmaxf(lmax, s);
  }
  red[t] = lmax; __syncthreads();
  for (int s = 128; s > 0; s >>= 1) { if (t < s) red[t] = fmaxf(red[t], red[t + s]); __syncthreads(); }
  float smax = red[0]; __syncthreads();
  float lsum = 0.f;
  for (int n = t; n < Lx; n += 256) { float p = __expf(sc[n] - smax); sc[n] = p; lsum += p; }
  red[t] = lsum; __syncthreads();
  for (int s = 128; s > 0; s >>= 1) { if (t < s) red[t] += red[t + s]; __syncthreads(); }
  float ssum = red[0]; __syncthreads();
  int dk = t & 63, part = t >> 6;                     // 4 partials per dk
  const float* Vb = Vf + (size_t)bh * Lx * DKx;
  float acc = 0.f;
  for (int n = part * 240; n < part * 240 + 240; ++n) acc += sc[n] * Vb[(size_t)n * DKx + dk];
  red[t] = acc; __syncthreads();
  if (t < DKx) {
    float o = (red[t] + red[t + 64] + red[t + 128] + red[t + 192]) / ssum;
    out_top[((size_t)bh * UTOP + u) * DKx + t] = o;
  }
}

// ---------------- 7: V mean over L ----------------
__global__ void k_vmean(const float* __restrict__ Vf, float* __restrict__ vmean) {
  int i = blockIdx.x * blockDim.x + threadIdx.x;
  if (i >= Bx * Hx * DKx) return;
  int bh = i >> 6, dk = i & 63;
  const float* Vb = Vf + (size_t)bh * Lx * DKx + dk;
  float acc = 0.f;
  for (int n = 0; n < Lx; ++n) acc += Vb[(size_t)n * DKx];
  vmean[i] = acc * (1.f / Lx);
}

// ---------------- 8: broadcast fill ----------------
__global__ void k_fill(const float* __restrict__ vmean, float* __restrict__ out) {
  int idx = blockIdx.x * blockDim.x + threadIdx.x;
  if (idx >= Bx * Lx * Dx) return;
  int d = idx & 511;
  int bl = idx >> 9;
  int b = bl / Lx;
  int h = d >> 6, dk = d & 63;
  out[idx] = vmean[(b * Hx + h) * DKx + dk];
}

// ---------------- 9: scatter top rows ----------------
__global__ void k_scatter(const float* __restrict__ out_top, const int* __restrict__ top_idx,
                          float* __restrict__ out) {
  int idx = blockIdx.x * blockDim.x + threadIdx.x;
  if (idx >= Bx * Hx * UTOP * DKx) return;
  int dk = idx & 63;
  int rest = idx >> 6;
  int u = rest % UTOP;
  int bh = rest / UTOP;
  int b = bh >> 3, h = bh & 7;
  int li = top_idx[bh * UTOP + u];
  out[((size_t)b * Lx + li) * Dx + h * DKx + dk] = out_top[idx];
}

// ---------------- launcher ----------------
extern "C" void kernel_launch(void* const* d_in, const int* in_sizes, int n_in,
                              void* d_out, int out_size, void* d_ws, size_t ws_size,
                              hipStream_t stream) {
  const float* values     = (const float*)d_in[0];
  const int*   times      = (const int*)d_in[1];
  const int*   sample_idx = (const int*)d_in[2];
  const float* w_val      = (const float*)d_in[3];
  const float* b_val      = (const float*)d_in[4];
  const float* pos_emb    = (const float*)d_in[5];
  const float* hour_tab   = (const float*)d_in[6];
  const float* week_tab   = (const float*)d_in[7];
  const float* month_tab  = (const float*)d_in[8];
  const float* conv_w     = (const float*)d_in[9];
  const float* conv_b     = (const float*)d_in[10];
  const float* wq         = (const float*)d_in[11];
  const float* bq         = (const float*)d_in[12];
  const float* wk         = (const float*)d_in[13];
  const float* bk         = (const float*)d_in[14];
  const float* wv         = (const float*)d_in[15];
  const float* bv         = (const float*)d_in[16];
  float* out = (float*)d_out;

  char* ws = (char*)d_ws;
  size_t off = 0;
  auto alloc = [&](size_t bytes) -> void* {
    void* p = ws + off;
    off += (bytes + 255) & ~(size_t)255;
    return p;
  };
  _Float16* convpack = (_Float16*)alloc((size_t)3 * Dx * Dx * 2);  // packed conv B frags
  _Float16* qkvpack  = (_Float16*)alloc((size_t)3 * Dx * Dx * 2);  // packed wq|wk|wv frags
  _Float16* emb16    = (_Float16*)alloc((size_t)MROWS * Dx * 2);
  _Float16* x16      = (_Float16*)alloc((size_t)MROWS * Dx * 2);
  _Float16* Q16      = (_Float16*)alloc((size_t)MROWS * Dx * 2);
  _Float16* K16      = (_Float16*)alloc((size_t)MROWS * Dx * 2);
  float*    Vf       = (float*)alloc((size_t)MROWS * Dx * 4);
  float*    Mbuf     = (float*)alloc((size_t)Bx * Hx * Lx * 4);
  int*      topidx   = (int*)alloc((size_t)Bx * Hx * UTOP * 4);
  float*    outtop   = (float*)alloc((size_t)Bx * Hx * UTOP * DKx * 4);
  float*    vmean    = (float*)alloc((size_t)Bx * Hx * DKx * 4);

  int n;
  n = NTILES * KC_CONV * 32;   // pack threads for conv weights (K=1536)
  k_pack_b<<<(n + 255) / 256, 256, 0, stream>>>(conv_w, 3 * Dx, convpack);
  n = NTILES * KC_QKV * 32;    // pack threads per qkv weight (K=512)
  k_pack_b<<<(n + 255) / 256, 256, 0, stream>>>(wq, Dx, qkvpack);
  k_pack_b<<<(n + 255) / 256, 256, 0, stream>>>(wk, Dx, qkvpack + Dx * Dx);
  k_pack_b<<<(n + 255) / 256, 256, 0, stream>>>(wv, Dx, qkvpack + 2 * Dx * Dx);

  n = MROWS * Dx;
  k_embed<<<(n + 255) / 256, 256, 0, stream>>>(values, times, w_val, b_val, pos_emb,
                                               hour_tab, week_tab, month_tab, emb16);

  k_conv_elu<<<MTILES, 256, 0, stream>>>(emb16, convpack, conv_b, x16);
  k_qkv<<<dim3(MTILES, 1, 3), 256, 0, stream>>>(x16, qkvpack, bq, bk, bv, Q16, K16, Vf);
  k_msparse<<<MTILES, 256, 0, stream>>>(Q16, K16, sample_idx, Mbuf);
  k_topk<<<Bx * Hx, 256, 0, stream>>>(Mbuf, topidx);
  k_topattn<<<Bx * Hx * UTOP, 256, 0, stream>>>(Q16, K16, Vf, topidx, outtop);
  k_vmean<<<(Bx * Hx * DKx + 255) / 256, 256, 0, stream>>>(Vf, vmean);
  n = Bx * Lx * Dx;
  k_fill<<<(n + 255) / 256, 256, 0, stream>>>(vmean, out);
  n = Bx * Hx * UTOP * DKx;
  k_scatter<<<(n + 255) / 256, 256, 0, stream>>>(outtop, topidx, out);
}
